// NodeEdgeConv_19232863552107
// MI455X (gfx1250) — matrix operand order repaired and checked by reading, not compiled, not run on hardware
//
#include <hip/hip_runtime.h>

// ---------------------------------------------------------------------------
// NodeEdgeConv for MI455X (gfx1250).
// Key identity: gather index == scatter index, so
//   out[d] = h[d] * ( segsum(V)[d] @ W + cnt[d]*b )
// -> segment-sum the 268 MB edge stream once (HBM-bound, NT loads, f32
//    atomics into L2-resident 2MB buffers), then tiny WMMA GEMMs per node.
// ---------------------------------------------------------------------------

typedef __bf16 bf16_t;
typedef __attribute__((ext_vector_type(16))) __bf16 v16bf;
typedef __attribute__((ext_vector_type(8)))  __bf16 v8bf;
typedef __attribute__((ext_vector_type(8)))  float  v8f;
typedef __attribute__((ext_vector_type(4)))  float  v4f;

#define D_DIM   128
#define M_DIM   64
#define N_NODES 8192
#define E_EDGES 524288

// ---------------------------------------------------------------------------
// WMMA wrapper: D = A(16x32 bf16) * B(32x16 bf16) + C(16x16 f32)
// ---------------------------------------------------------------------------
__device__ __forceinline__ v8f wmma_bf16(v16bf a, v16bf b, v8f c) {
  return __builtin_amdgcn_wmma_f32_16x16x32_bf16(
      /*neg_a=*/false, a, /*neg_b=*/false, b,
      /*c_mod=*/(short)0, c, /*reuse_a=*/false, /*reuse_b=*/false);
}

// A-fragment (16x32 bf16) from f32 row-major A[ld], converting on the fly.
// ISA layout: lanes 0-15 -> M=lane, elems 0..7 = K 0..7, elems 8..15 = K 16..23
//             lanes16-31 -> M=lane-16, elems 0..7 = K 8..15, 8..15 = K 24..31
__device__ __forceinline__ v16bf load_a_frag_f32(const float* __restrict__ A, int ld,
                                                 int r0, int k0, int lane) {
  const int m  = lane & 15;
  const int rh = lane >> 4;
  const float* p = A + (size_t)(r0 + m) * (size_t)ld + (size_t)(k0 + rh * 8);
  const float4 f0 = *reinterpret_cast<const float4*>(p);
  const float4 f1 = *reinterpret_cast<const float4*>(p + 4);
  const float4 f2 = *reinterpret_cast<const float4*>(p + 16);
  const float4 f3 = *reinterpret_cast<const float4*>(p + 20);
  v16bf a;
  a[0]  = (bf16_t)f0.x; a[1]  = (bf16_t)f0.y; a[2]  = (bf16_t)f0.z; a[3]  = (bf16_t)f0.w;
  a[4]  = (bf16_t)f1.x; a[5]  = (bf16_t)f1.y; a[6]  = (bf16_t)f1.z; a[7]  = (bf16_t)f1.w;
  a[8]  = (bf16_t)f2.x; a[9]  = (bf16_t)f2.y; a[10] = (bf16_t)f2.z; a[11] = (bf16_t)f2.w;
  a[12] = (bf16_t)f3.x; a[13] = (bf16_t)f3.y; a[14] = (bf16_t)f3.z; a[15] = (bf16_t)f3.w;
  return a;
}

// B-fragment (32x16 bf16) from pre-transposed bf16 weight Wt[N][K].
// lane = N (mod 16); lanes 0-15 hold K=k0..k0+7,k0+16..k0+23, lanes 16-31 K=k0+8..k0+15,k0+24..k0+31.
__device__ __forceinline__ v16bf load_b_frag(const bf16_t* __restrict__ Wt, int K,
                                             int c0, int k0, int lane) {
  const int n  = lane & 15;
  const int kb = k0 + (lane >> 4) * 8;
  const bf16_t* q = Wt + (size_t)(c0 + n) * (size_t)K + kb;
  v8bf lo = *reinterpret_cast<const v8bf*>(q);
  v8bf hi = *reinterpret_cast<const v8bf*>(q + 16);
  return __builtin_shufflevector(lo, hi, 0, 1, 2, 3, 4, 5, 6, 7, 8, 9, 10, 11, 12, 13, 14, 15);
}

// ---------------------------------------------------------------------------
// Prep: W f32 row-major [K][N]  ->  Wt bf16 transposed [N][K]
// ---------------------------------------------------------------------------
__global__ void convert_wt_kernel(const float* __restrict__ W, bf16_t* __restrict__ Wt,
                                  int K, int N) {
  const int i = blockIdx.x * blockDim.x + threadIdx.x;
  if (i < K * N) {
    const int k = i / N, n = i % N;
    Wt[(size_t)n * K + k] = (bf16_t)W[i];
  }
}

// ---------------------------------------------------------------------------
// Segment-sum of the edge value stream (the only HBM-bound stage):
//   Vsum[idx[e]][:] += V[e][:]   ;   cnt[idx[e]] += 1
// 16 threads per edge, one float4 each. NT loads: 268 MB one-shot stream must
// not evict the L2-resident Vsum / h / weight working set.
// ---------------------------------------------------------------------------
__global__ __launch_bounds__(256) void segsum_kernel(
    const float* __restrict__ V, const int* __restrict__ idx,
    float* __restrict__ Vsum, float* __restrict__ cnt) {
  const int gid = blockIdx.x * 256 + threadIdx.x;
  const int e   = gid >> 4;   // edge id
  const int c4  = gid & 15;   // float4 chunk within the 64-wide row
  if ((threadIdx.x & 31) == 0)  // hint the next streaming tile
    __builtin_prefetch(V + (size_t)(gid + 256) * 4, 0, 0);
  const v4f v = __builtin_nontemporal_load(
      reinterpret_cast<const v4f*>(V + (size_t)e * M_DIM + c4 * 4));
  const int d = idx[e];
  float* dst = Vsum + (size_t)d * M_DIM + c4 * 4;
  atomicAdd(dst + 0, v[0]);
  atomicAdd(dst + 1, v[1]);
  atomicAdd(dst + 2, v[2]);
  atomicAdd(dst + 3, v[3]);
  if (c4 == 0) atomicAdd(cnt + d, 1.0f);
}

// ---------------------------------------------------------------------------
// Node transform: H[N_NODES][128] = X @ W + b    (K = 128)
// Block = 256 threads = 8 waves -> 16 rows x 128 cols per block.
// ---------------------------------------------------------------------------
__global__ __launch_bounds__(256) void node_gemm_kernel(
    const float* __restrict__ X, const bf16_t* __restrict__ Wt,
    const float* __restrict__ bias, float* __restrict__ H) {
  const int lane = threadIdx.x & 31;
  const int wv   = threadIdx.x >> 5;
  const int r0   = blockIdx.x * 16;
  const int c0   = wv * 16;
  v8f acc = {};
#pragma unroll
  for (int k0 = 0; k0 < D_DIM; k0 += 32) {
    v16bf a = load_a_frag_f32(X, D_DIM, r0, k0, lane);
    v16bf b = load_b_frag(Wt, D_DIM, c0, k0, lane);
    acc = wmma_bf16(a, b, acc);
  }
  const int n  = lane & 15;
  const int rh = lane >> 4;
  const float bc = bias[c0 + n];
#pragma unroll
  for (int i = 0; i < 8; ++i)
    H[(size_t)(r0 + rh * 8 + i) * D_DIM + c0 + n] = acc[i] + bc;
}

// ---------------------------------------------------------------------------
// Per-node message:  Acc = H * (Vsum @ Wm + cnt * bm)     (K = 64)
// ---------------------------------------------------------------------------
__global__ __launch_bounds__(256) void msg_node_kernel(
    const float* __restrict__ Vsum, const float* __restrict__ cnt,
    const bf16_t* __restrict__ Wt, const float* __restrict__ bias,
    const float* __restrict__ H, float* __restrict__ Acc) {
  const int lane = threadIdx.x & 31;
  const int wv   = threadIdx.x >> 5;
  const int r0   = blockIdx.x * 16;
  const int c0   = wv * 16;
  v8f acc = {};
#pragma unroll
  for (int k0 = 0; k0 < M_DIM; k0 += 32) {
    v16bf a = load_a_frag_f32(Vsum, M_DIM, r0, k0, lane);
    v16bf b = load_b_frag(Wt, M_DIM, c0, k0, lane);
    acc = wmma_bf16(a, b, acc);
  }
  const int n  = lane & 15;
  const int rh = lane >> 4;
  const float bc = bias[c0 + n];
#pragma unroll
  for (int i = 0; i < 8; ++i) {
    const int r = r0 + rh * 8 + i;
    const size_t o = (size_t)r * D_DIM + c0 + n;
    Acc[o] = H[o] * (acc[i] + cnt[r] * bc);
  }
}

// ---------------------------------------------------------------------------
// Fused apply: Out = Xres + LN(Acc @ W1 + b1; g, beta) @ W2 + b2
// GEMM1 -> LDS -> block LayerNorm -> bf16 LDS -> GEMM2 -> bias + residual.
// ---------------------------------------------------------------------------
__global__ __launch_bounds__(256) void apply_kernel(
    const float* __restrict__ Acc, const float* __restrict__ Xres,
    const bf16_t* __restrict__ W1t, const float* __restrict__ b1,
    const float* __restrict__ g, const float* __restrict__ beta,
    const bf16_t* __restrict__ W2t, const float* __restrict__ b2,
    float* __restrict__ Out) {
  __shared__ float  H1[16][D_DIM + 4];
  __shared__ bf16_t H1n[16][D_DIM];
  __shared__ float  redS[16][16];
  __shared__ float  redQ[16][16];
  __shared__ float  muS[16], rsS[16];

  const int lane = threadIdx.x & 31;
  const int wv   = threadIdx.x >> 5;
  const int r0   = blockIdx.x * 16;
  const int c0   = wv * 16;
  const int n    = lane & 15;
  const int rh   = lane >> 4;

  // --- GEMM1: H1 = Acc @ W1 + b1 ---
  {
    v8f acc = {};
#pragma unroll
    for (int k0 = 0; k0 < D_DIM; k0 += 32) {
      v16bf a = load_a_frag_f32(Acc, D_DIM, r0, k0, lane);
      v16bf b = load_b_frag(W1t, D_DIM, c0, k0, lane);
      acc = wmma_bf16(a, b, acc);
    }
    const float bc = b1[c0 + n];
#pragma unroll
    for (int i = 0; i < 8; ++i)
      H1[rh * 8 + i][c0 + n] = acc[i] + bc;
  }
  __syncthreads();

  // --- LayerNorm over each 128-wide row ---
  {
    const int row = threadIdx.x >> 4;
    const int seg = threadIdx.x & 15;
    float s = 0.f, q = 0.f;
#pragma unroll
    for (int j = 0; j < 8; ++j) {
      const float v = H1[row][seg * 8 + j];
      s += v;
      q += v * v;
    }
    redS[row][seg] = s;
    redQ[row][seg] = q;
  }
  __syncthreads();
  if (threadIdx.x < 16) {
    float s = 0.f, q = 0.f;
#pragma unroll
    for (int j = 0; j < 16; ++j) { s += redS[threadIdx.x][j]; q += redQ[threadIdx.x][j]; }
    const float mu  = s * (1.0f / D_DIM);
    const float var = q * (1.0f / D_DIM) - mu * mu;
    muS[threadIdx.x] = mu;
    rsS[threadIdx.x] = rsqrtf(var + 1e-5f);
  }
  __syncthreads();
  {
    const int row = threadIdx.x >> 4;
    const int seg = threadIdx.x & 15;
    const float mu = muS[row], rs = rsS[row];
#pragma unroll
    for (int j = 0; j < 8; ++j) {
      const int c = seg * 8 + j;
      H1n[row][c] = (bf16_t)((H1[row][c] - mu) * rs * g[c] + beta[c]);
    }
  }
  __syncthreads();

  // --- GEMM2: Out = Xres + H1n @ W2 + b2 (A-fragments from LDS) ---
  {
    const int m = lane & 15;
    v8f acc = {};
#pragma unroll
    for (int k0 = 0; k0 < D_DIM; k0 += 32) {
      v8bf lo = *reinterpret_cast<const v8bf*>(&H1n[m][k0 + rh * 8]);
      v8bf hi = *reinterpret_cast<const v8bf*>(&H1n[m][k0 + 16 + rh * 8]);
      v16bf a = __builtin_shufflevector(lo, hi, 0, 1, 2, 3, 4, 5, 6, 7,
                                        8, 9, 10, 11, 12, 13, 14, 15);
      v16bf b = load_b_frag(W2t, D_DIM, c0, k0, lane);
      acc = wmma_bf16(a, b, acc);
    }
    const float bc = b2[c0 + n];
#pragma unroll
    for (int i = 0; i < 8; ++i) {
      const size_t o = (size_t)(r0 + rh * 8 + i) * D_DIM + c0 + n;
      Out[o] = Xres[o] + acc[i] + bc;
    }
  }
}

// ---------------------------------------------------------------------------
// Host launcher
// ---------------------------------------------------------------------------
extern "C" void kernel_launch(void* const* d_in, const int* in_sizes, int n_in,
                              void* d_out, int out_size, void* d_ws, size_t ws_size,
                              hipStream_t stream) {
  (void)in_sizes; (void)n_in; (void)out_size; (void)ws_size;

  const float* src_embed = (const float*)d_in[0];
  const float* dst_embed = (const float*)d_in[1];
  const float* v_s2d     = (const float*)d_in[2];
  const float* v_d2s     = (const float*)d_in[3];
  const int*   e_s2d     = (const int*)d_in[4];
  const int*   e_d2s     = (const int*)d_in[5];
  const float* W_src = (const float*)d_in[6];
  const float* b_src = (const float*)d_in[7];
  const float* W_dst = (const float*)d_in[8];
  const float* b_dst = (const float*)d_in[9];
  const float* W_sm  = (const float*)d_in[10];
  const float* b_sm  = (const float*)d_in[11];
  const float* W_dm  = (const float*)d_in[12];
  const float* b_dm  = (const float*)d_in[13];
  const float* row_W1 = (const float*)d_in[14];
  const float* row_b1 = (const float*)d_in[15];
  const float* row_g  = (const float*)d_in[16];
  const float* row_be = (const float*)d_in[17];
  const float* row_W2 = (const float*)d_in[18];
  const float* row_b2 = (const float*)d_in[19];
  const float* col_W1 = (const float*)d_in[20];
  const float* col_b1 = (const float*)d_in[21];
  const float* col_g  = (const float*)d_in[22];
  const float* col_be = (const float*)d_in[23];
  const float* col_W2 = (const float*)d_in[24];
  const float* col_b2 = (const float*)d_in[25];

  float* out_row = (float*)d_out;                              // [8192,128]
  float* out_col = (float*)d_out + (size_t)N_NODES * D_DIM;    // [8192,128]

  // Workspace carve-out (256B aligned slots)
  char* ws = (char*)d_ws;
  size_t off = 0;
  auto wsalloc = [&](size_t bytes) -> void* {
    void* p = ws + off;
    off += (bytes + 255) & ~(size_t)255;
    return p;
  };
  const size_t NODE_BYTES = (size_t)N_NODES * D_DIM * sizeof(float); // 4 MB
  const size_t VSUM_BYTES = (size_t)N_NODES * M_DIM * sizeof(float); // 2 MB
  const size_t CNT_BYTES  = (size_t)N_NODES * sizeof(float);         // 32 KB
  float*  h_src   = (float*)wsalloc(NODE_BYTES);
  float*  h_dst   = (float*)wsalloc(NODE_BYTES);
  float*  acc_src = (float*)wsalloc(NODE_BYTES);
  float*  acc_dst = (float*)wsalloc(NODE_BYTES);
  float*  vsum_s  = (float*)wsalloc(VSUM_BYTES);   // d2s edges -> src nodes
  float*  vsum_d  = (float*)wsalloc(VSUM_BYTES);   // s2d edges -> dst nodes
  float*  cnt_s   = (float*)wsalloc(CNT_BYTES);
  float*  cnt_d   = (float*)wsalloc(CNT_BYTES);
  bf16_t* wt_src  = (bf16_t*)wsalloc((size_t)D_DIM * D_DIM * 2);
  bf16_t* wt_dst  = (bf16_t*)wsalloc((size_t)D_DIM * D_DIM * 2);
  bf16_t* wt_sm   = (bf16_t*)wsalloc((size_t)M_DIM * D_DIM * 2);
  bf16_t* wt_dm   = (bf16_t*)wsalloc((size_t)M_DIM * D_DIM * 2);
  bf16_t* wt_rw1  = (bf16_t*)wsalloc((size_t)D_DIM * D_DIM * 2);
  bf16_t* wt_rw2  = (bf16_t*)wsalloc((size_t)D_DIM * D_DIM * 2);
  bf16_t* wt_cw1  = (bf16_t*)wsalloc((size_t)D_DIM * D_DIM * 2);
  bf16_t* wt_cw2  = (bf16_t*)wsalloc((size_t)D_DIM * D_DIM * 2);

  // Zero the segment-sum accumulators (capture-legal, stream-ordered)
  hipMemsetAsync(vsum_s, 0, VSUM_BYTES, stream);
  hipMemsetAsync(vsum_d, 0, VSUM_BYTES, stream);
  hipMemsetAsync(cnt_s, 0, CNT_BYTES, stream);
  hipMemsetAsync(cnt_d, 0, CNT_BYTES, stream);

  // Weight transpose + bf16 convert
  convert_wt_kernel<<<64, 256, 0, stream>>>(W_src,  wt_src, D_DIM, D_DIM);
  convert_wt_kernel<<<64, 256, 0, stream>>>(W_dst,  wt_dst, D_DIM, D_DIM);
  convert_wt_kernel<<<32, 256, 0, stream>>>(W_sm,   wt_sm,  M_DIM, D_DIM);
  convert_wt_kernel<<<32, 256, 0, stream>>>(W_dm,   wt_dm,  M_DIM, D_DIM);
  convert_wt_kernel<<<64, 256, 0, stream>>>(row_W1, wt_rw1, D_DIM, D_DIM);
  convert_wt_kernel<<<64, 256, 0, stream>>>(row_W2, wt_rw2, D_DIM, D_DIM);
  convert_wt_kernel<<<64, 256, 0, stream>>>(col_W1, wt_cw1, D_DIM, D_DIM);
  convert_wt_kernel<<<64, 256, 0, stream>>>(col_W2, wt_cw2, D_DIM, D_DIM);

  // Segment-sum the edge streams (HBM-bound stage: one pass over 268 MB)
  segsum_kernel<<<(E_EDGES * 16) / 256, 256, 0, stream>>>(v_s2d, e_s2d, vsum_d, cnt_d);
  segsum_kernel<<<(E_EDGES * 16) / 256, 256, 0, stream>>>(v_d2s, e_d2s, vsum_s, cnt_s);

  // Node transforms: h_src, h_dst
  node_gemm_kernel<<<N_NODES / 16, 256, 0, stream>>>(src_embed, wt_src, b_src, h_src);
  node_gemm_kernel<<<N_NODES / 16, 256, 0, stream>>>(dst_embed, wt_dst, b_dst, h_dst);

  // Per-node aggregated message:  acc = h * (Vsum @ Wm + cnt * bm)
  msg_node_kernel<<<N_NODES / 16, 256, 0, stream>>>(vsum_d, cnt_d, wt_sm, b_sm, h_dst, acc_dst);
  msg_node_kernel<<<N_NODES / 16, 256, 0, stream>>>(vsum_s, cnt_s, wt_dm, b_dm, h_src, acc_src);

  // Residual apply (row then col, matching reference return order)
  apply_kernel<<<N_NODES / 16, 256, 0, stream>>>(
      acc_src, src_embed, wt_rw1, row_b1, row_g, row_be, wt_rw2, row_b2, out_row);
  apply_kernel<<<N_NODES / 16, 256, 0, stream>>>(
      acc_dst, dst_embed, wt_cw1, col_b1, col_g, col_be, wt_cw2, col_b2, out_col);
}
